// GAT_54150947668618
// MI455X (gfx1250) — matrix operand order repaired and checked by reading, not compiled, not run on hardware
//
#include <hip/hip_runtime.h>
#include <hip/hip_bf16.h>
#include <stdint.h>

// ---------------------------------------------------------------------------
// GAT forward on MI455X (gfx1250): bf16 WMMA + flash-style fused softmax with
// barrier-free, wave-private, async double-buffered adj tiles.
//   N=8192, NFEAT=512, NHID=64, NCLASS=16, NHEAD=4
// ---------------------------------------------------------------------------

#define GN     8192
#define GF     512
#define GD     64
#define GH     4
#define GC     16
#define GNEG   (-9e15f)
#define GALPHA 0.1f
#define AW     36      // padded adj-tile row stride in floats (144B, 16B-aligned)

typedef __attribute__((ext_vector_type(16))) __bf16 v16bf;
typedef __attribute__((ext_vector_type(8)))  __bf16 v8bf;
typedef __attribute__((ext_vector_type(8)))  float  v8f;
typedef __attribute__((ext_vector_type(4)))  int    v4i;

// address-space-qualified pointee types for the async DMA builtin
typedef __attribute__((address_space(1))) v4i as1_v4i;   // global
typedef __attribute__((address_space(3))) v4i as3_v4i;   // LDS

#if defined(__HIP_DEVICE_COMPILE__) && \
    __has_builtin(__builtin_amdgcn_global_load_async_to_lds_b128) && \
    __has_builtin(__builtin_amdgcn_s_wait_asynccnt)
#define HAVE_ASYNC 1
#else
#define HAVE_ASYNC 0
#endif

__device__ __forceinline__ v16bf cat8(v8bf lo, v8bf hi) {
  return __builtin_shufflevector(lo, hi, 0,1,2,3,4,5,6,7,8,9,10,11,12,13,14,15);
}

__device__ __forceinline__ v8f wmma_bf16(v16bf a, v16bf b, v8f c) {
  // D = A(16x32) * B(32x16) + C, f32 accumulate
  return __builtin_amdgcn_wmma_f32_16x16x32_bf16(false, a, false, b,
                                                 (short)0, c, false, false);
}

#if HAVE_ASYNC
__device__ __forceinline__ void async_ld16(const void* g, void* l) {
  // global (AS1) -> LDS (AS3) 16-byte async DMA, tracked by ASYNCcnt.
  // AS1 pointer: same bit pattern as the flat pointer.
  // AS3 pointer: low 32 bits of the flat LDS address (ISA 10.2 aperture map).
  as1_v4i* gp = (as1_v4i*)(uintptr_t)g;
  as3_v4i* lp = (as3_v4i*)(unsigned int)(uintptr_t)l;
  __builtin_amdgcn_global_load_async_to_lds_b128(gp, lp, 0, 0);
}
#endif

// ---------------------------------------------------------------------------
// Small conversion / transpose kernels (negligible cost, feed WMMA layouts)
// ---------------------------------------------------------------------------
__global__ void __launch_bounds__(256)
cvt_f32_bf16_kernel(const float* __restrict__ in, __bf16* __restrict__ outb, int n) {
  int i = blockIdx.x * blockDim.x + threadIdx.x;
  if (i < n) outb[i] = (__bf16)in[i];
}

// W [H][F][D] f32 -> WT [H][D][F] bf16 (d-major so B operands are contiguous in k)
__global__ void __launch_bounds__(256)
transpose_W_kernel(const float* __restrict__ W, __bf16* __restrict__ WT) {
  int i = blockIdx.x * blockDim.x + threadIdx.x;
  if (i >= GH * GF * GD) return;
  int d = i % GD;
  int k = (i / GD) % GF;
  int h = i / (GD * GF);
  WT[((size_t)h * GD + d) * GF + k] = (__bf16)W[i];
}

// W_out [256][16] f32 -> WoutT [16][256] bf16
__global__ void __launch_bounds__(256)
transpose_Wout_kernel(const float* __restrict__ Wo, __bf16* __restrict__ WoT) {
  int i = blockIdx.x * blockDim.x + threadIdx.x;
  if (i >= (GH * GD) * GC) return;
  int c = i % GC;
  int k = i / GC;
  WoT[(size_t)c * (GH * GD) + k] = (__bf16)Wo[i];
}

// ---------------------------------------------------------------------------
// Layer-1 feature GEMM:  WhT[h][d][i] = (x @ W[h])[i][d]    (bf16 WMMA)
// ---------------------------------------------------------------------------
__global__ void __launch_bounds__(256)
gemm1_kernel(const __bf16* __restrict__ xbf, const __bf16* __restrict__ WT,
             __bf16* __restrict__ WhT) {
  const int w = threadIdx.x >> 5, lane = threadIdx.x & 31;
  const int hi = lane >> 4, lo = lane & 15;
  const int h = blockIdx.y;
  const int rowbase = blockIdx.x * 128 + w * 16;

  v8f acc[4] = {};
  const __bf16* arow = xbf + (size_t)(rowbase + lo) * GF;
  const __bf16* wt   = WT + (size_t)h * GD * GF;

  for (int k0 = 0; k0 < GF; k0 += 32) {
    v16bf A = cat8(*(const v8bf*)(arow + k0 + hi * 8),
                   *(const v8bf*)(arow + k0 + hi * 8 + 16));
    const int kk = k0 + hi * 8;
#pragma unroll
    for (int g = 0; g < 4; ++g) {
      const __bf16* bp = wt + (size_t)(g * 16 + lo) * GF + kk;
      v16bf B = cat8(*(const v8bf*)bp, *(const v8bf*)(bp + 16));
      acc[g] = wmma_bf16(A, B, acc[g]);
    }
  }
#pragma unroll
  for (int g = 0; g < 4; ++g)
#pragma unroll
    for (int r = 0; r < 8; ++r) {
      int rowg = rowbase + r + hi * 8;
      int d    = g * 16 + lo;
      WhT[(size_t)(h * GD + d) * GN + rowg] = (__bf16)acc[g][r];
    }
}

// ---------------------------------------------------------------------------
// f1/f2 vectors (layer 1): f1t[i][h] = Wh[i,:].a[h][:64], f2t = .a[h][64:]
// ---------------------------------------------------------------------------
__global__ void __launch_bounds__(256)
fvec1_kernel(const __bf16* __restrict__ WhT, const float* __restrict__ a,
             float* __restrict__ f1t, float* __restrict__ f2t) {
  int i = blockIdx.x * blockDim.x + threadIdx.x;
  if (i >= GN) return;
#pragma unroll
  for (int h = 0; h < GH; ++h) {
    float s1 = 0.f, s2 = 0.f;
    const float* ah = a + h * (2 * GD);
#pragma unroll 8
    for (int d = 0; d < GD; ++d) {
      float wv = (float)WhT[(size_t)(h * GD + d) * GN + i];
      s1 += wv * ah[d];
      s2 += wv * ah[GD + d];
    }
    f1t[i * GH + h] = s1;
    f2t[i * GH + h] = s2;
  }
}

// ---------------------------------------------------------------------------
// Row-max pass: m[i][h] = max( lrelu(f1[i][h] + max_{j in nbr(i)} f2[j][h]), NEG )
// ---------------------------------------------------------------------------
template <int NH>
__global__ void __launch_bounds__(256)
rowmax_kernel(const float* __restrict__ adj, const float* __restrict__ f1t,
              const float* __restrict__ f2t, float* __restrict__ mout) {
  const int w = threadIdx.x >> 5, lane = threadIdx.x & 31;
  const int row = blockIdx.x * 8 + w;
  const float* ar = adj + (size_t)row * GN;

  float mx[NH];
#pragma unroll
  for (int h = 0; h < NH; ++h) mx[h] = -3.0e38f;

  for (int j = lane * 4; j < GN; j += 128) {
    float4 av = *(const float4*)(ar + j);
    float a4[4] = {av.x, av.y, av.z, av.w};
#pragma unroll
    for (int u = 0; u < 4; ++u) {
      if (a4[u] > 0.f) {
#pragma unroll
        for (int h = 0; h < NH; ++h)
          mx[h] = fmaxf(mx[h], f2t[(size_t)(j + u) * NH + h]);
      }
    }
  }
#pragma unroll
  for (int h = 0; h < NH; ++h) {
#pragma unroll
    for (int sh = 1; sh < 32; sh <<= 1)
      mx[h] = fmaxf(mx[h], __shfl_xor(mx[h], sh));
  }
  if (lane == 0) {
#pragma unroll
    for (int h = 0; h < NH; ++h) {
      float t = f1t[(size_t)row * NH + h] + mx[h];
      float e = t > 0.f ? t : GALPHA * t;
      mout[(size_t)row * NH + h] = fmaxf(e, GNEG);  // empty row -> NEG -> uniform
    }
  }
}

// ---------------------------------------------------------------------------
// Layer-1 fused attention+aggregate (flash-style), barrier-free:
// each wave owns a 16-row tile and processes ALL 4 heads, so its adj/f2 LDS
// tiles are wave-private and double-buffered via async DMA (no __syncthreads).
//   out[i][h*64+d] = elu( (sum_j p_ij Wh[j][d]) / (sum_j p_ij) )
//   p_ij = exp( (adj>0 ? lrelu(f1_i+f2_j) : NEG) - m_i )
// grid N/128; 8 waves/block; adj read exactly once.
// ---------------------------------------------------------------------------
__global__ void __launch_bounds__(256)
attn1_kernel(const float* __restrict__ adj, const __bf16* __restrict__ WhT,
             const float* __restrict__ f1t, const float* __restrict__ f2t,
             const float* __restrict__ m1, __bf16* __restrict__ hbf) {
  __shared__ __align__(16) float adjS[8][2][16][AW];   // per-wave double buffer
  __shared__ __align__(16) float f2S[8][2][32 * GH];

  const int w = threadIdx.x >> 5, lane = threadIdx.x & 31;
  const int hi = lane >> 4, lo = lane & 15;
  const int rowbase = blockIdx.x * 128 + w * 16;

  float f1r[GH], mr[GH];
#pragma unroll
  for (int h = 0; h < GH; ++h) {
    f1r[h] = f1t[(size_t)(rowbase + lo) * GH + h];
    mr[h]  = m1[(size_t)(rowbase + lo) * GH + h];
  }

  v8f acc[GH][4] = {};
  float ssum[GH] = {0.f, 0.f, 0.f, 0.f};

  const int ar = lane >> 3;        // row-in-group-of-4
  const int ac = (lane & 7) * 4;   // col (floats)
  const float* adj_base = adj + (size_t)rowbase * GN;

  // 5 load ops per chunk per wave: 4x adj(16B/lane) + 1x f2(16B/lane)
  auto issue_tile = [&](int jb, int buf) {
#if HAVE_ASYNC
#pragma unroll
    for (int u = 0; u < 4; ++u) {
      int r16 = u * 4 + ar;
      async_ld16(adj_base + (size_t)r16 * GN + jb + ac, &adjS[w][buf][r16][ac]);
    }
    async_ld16(f2t + (size_t)jb * GH + lane * 4, &f2S[w][buf][lane * 4]);
#else
#pragma unroll
    for (int u = 0; u < 4; ++u) {
      int r16 = u * 4 + ar;
      *(float4*)&adjS[w][buf][r16][ac] =
          *(const float4*)(adj_base + (size_t)r16 * GN + jb + ac);
    }
    *(float4*)&f2S[w][buf][lane * 4] =
        *(const float4*)(f2t + (size_t)jb * GH + lane * 4);
#endif
  };

  issue_tile(0, 0);
  for (int jb = 0; jb < GN; jb += 32) {
    const int buf = (jb >> 5) & 1;
    const bool more = (jb + 32) < GN;
    if (more) issue_tile(jb + 32, buf ^ 1);
#if HAVE_ASYNC
    if (more) __builtin_amdgcn_s_wait_asynccnt(5);   // current chunk landed
    else      __builtin_amdgcn_s_wait_asynccnt(0);
#endif

    // adj values for this lane's A slice: shared across all 4 heads
    float av[16];
#pragma unroll
    for (int e = 0; e < 16; ++e) {
      int jj = hi * 8 + (e < 8 ? e : e + 8);
      av[e] = adjS[w][buf][lo][jj];
    }

#pragma unroll
    for (int h = 0; h < GH; ++h) {
      v16bf A;
      float ps = 0.f;
#pragma unroll
      for (int e = 0; e < 16; ++e) {
        int jj = hi * 8 + (e < 8 ? e : e + 8);
        float tt = f1r[h] + f2S[w][buf][jj * GH + h];
        float ee = av[e] > 0.f ? (tt > 0.f ? tt : GALPHA * tt) : GNEG;
        float p  = __expf(ee - mr[h]);
        ps += p;
        A[e] = (__bf16)p;
      }
      ssum[h] += ps;
#pragma unroll
      for (int g = 0; g < 4; ++g) {
        const __bf16* bp = WhT + (size_t)(h * GD + g * 16 + lo) * GN + jb + hi * 8;
        v16bf B = cat8(*(const v8bf*)bp, *(const v8bf*)(bp + 16));
        acc[h][g] = wmma_bf16(A, B, acc[h][g]);
      }
    }
  }

#pragma unroll
  for (int h = 0; h < GH; ++h) {
    float sf = ssum[h] + __shfl_xor(ssum[h], 16);  // denom for row (lane&15)
#pragma unroll
    for (int r = 0; r < 8; ++r) {
      int rr = r + hi * 8;
      float s = __shfl(sf, rr);
      int rowg = rowbase + rr;
#pragma unroll
      for (int g = 0; g < 4; ++g) {
        float v = acc[h][g][r] / s;
        v = v > 0.f ? v : (__expf(v) - 1.f);       // elu
        hbf[(size_t)rowg * (GH * GD) + h * GD + g * 16 + lo] = (__bf16)v;
      }
    }
  }
}

// ---------------------------------------------------------------------------
// Layer-2 feature GEMM + f-vectors:
//   Wh2 = h @ W_out (8192x256x16), store Wh2T bf16 [16][N]; f1_2/f2_2 = Wh2@a_out
// ---------------------------------------------------------------------------
__global__ void __launch_bounds__(256)
gemm2_kernel(const __bf16* __restrict__ hbf, const __bf16* __restrict__ WoutT,
             const float* __restrict__ aout, __bf16* __restrict__ Wh2T,
             float* __restrict__ f12, float* __restrict__ f22) {
  const int w = threadIdx.x >> 5, lane = threadIdx.x & 31;
  const int hi = lane >> 4, lo = lane & 15;
  const int rowbase = blockIdx.x * 128 + w * 16;

  v8f acc = {};
  const __bf16* arow = hbf + (size_t)(rowbase + lo) * (GH * GD);
  for (int k0 = 0; k0 < GH * GD; k0 += 32) {
    v16bf A = cat8(*(const v8bf*)(arow + k0 + hi * 8),
                   *(const v8bf*)(arow + k0 + hi * 8 + 16));
    const __bf16* bp = WoutT + (size_t)lo * (GH * GD) + k0 + hi * 8;
    v16bf B = cat8(*(const v8bf*)bp, *(const v8bf*)(bp + 16));
    acc = wmma_bf16(A, B, acc);
  }

  float a1 = aout[lo], a2 = aout[GC + lo];
#pragma unroll
  for (int r = 0; r < 8; ++r) {
    float v = acc[r];
    int rowg = rowbase + r + hi * 8;
    Wh2T[(size_t)lo * GN + rowg] = (__bf16)v;
    float p1 = v * a1, p2 = v * a2;
#pragma unroll
    for (int m = 1; m < 16; m <<= 1) {
      p1 += __shfl_xor(p1, m);
      p2 += __shfl_xor(p2, m);
    }
    if (lo == 0) { f12[rowg] = p1; f22[rowg] = p2; }
  }
}

// ---------------------------------------------------------------------------
// Layer-2 fused attention+aggregate + elu + log_softmax, barrier-free,
// wave-private async double-buffered tiles. grid N/128; 8 waves x 16 rows.
// ---------------------------------------------------------------------------
__global__ void __launch_bounds__(256)
attn2_kernel(const float* __restrict__ adj, const __bf16* __restrict__ Wh2T,
             const float* __restrict__ f12, const float* __restrict__ f22,
             const float* __restrict__ m2, float* __restrict__ out) {
  __shared__ __align__(16) float adjS[8][2][16][AW];
  __shared__ __align__(16) float f2S[8][2][32];

  const int w = threadIdx.x >> 5, lane = threadIdx.x & 31;
  const int hi = lane >> 4, lo = lane & 15;
  const int rowbase = blockIdx.x * 128 + w * 16;
  const float f1r = f12[rowbase + lo], mr = m2[rowbase + lo];

  v8f acc = {};
  float ssum = 0.f;

  const int ar = lane >> 3;
  const int ac = (lane & 7) * 4;
  const float* adj_base = adj + (size_t)rowbase * GN;

  auto issue_tile = [&](int jb, int buf) {
#if HAVE_ASYNC
#pragma unroll
    for (int u = 0; u < 4; ++u) {
      int r16 = u * 4 + ar;
      async_ld16(adj_base + (size_t)r16 * GN + jb + ac, &adjS[w][buf][r16][ac]);
    }
    // 32-float chunk: lanes 8..31 duplicate lanes 0..7 (identical data, benign)
    async_ld16(f22 + jb + (lane & 7) * 4, &f2S[w][buf][(lane & 7) * 4]);
#else
#pragma unroll
    for (int u = 0; u < 4; ++u) {
      int r16 = u * 4 + ar;
      *(float4*)&adjS[w][buf][r16][ac] =
          *(const float4*)(adj_base + (size_t)r16 * GN + jb + ac);
    }
    if (lane < 8)
      *(float4*)&f2S[w][buf][lane * 4] = *(const float4*)(f22 + jb + lane * 4);
#endif
  };

  issue_tile(0, 0);
  for (int jb = 0; jb < GN; jb += 32) {
    const int buf = (jb >> 5) & 1;
    const bool more = (jb + 32) < GN;
    if (more) issue_tile(jb + 32, buf ^ 1);
#if HAVE_ASYNC
    if (more) __builtin_amdgcn_s_wait_asynccnt(5);
    else      __builtin_amdgcn_s_wait_asynccnt(0);
#endif

    v16bf A;
    float ps = 0.f;
#pragma unroll
    for (int e = 0; e < 16; ++e) {
      int jj = hi * 8 + (e < 8 ? e : e + 8);
      float av = adjS[w][buf][lo][jj];
      float tt = f1r + f2S[w][buf][jj];
      float ee = av > 0.f ? (tt > 0.f ? tt : GALPHA * tt) : GNEG;
      float p  = __expf(ee - mr);
      ps += p;
      A[e] = (__bf16)p;
    }
    ssum += ps;

    const __bf16* bp = Wh2T + (size_t)lo * GN + jb + hi * 8;
    v16bf B = cat8(*(const v8bf*)bp, *(const v8bf*)(bp + 16));
    acc = wmma_bf16(A, B, acc);
  }

  float sf = ssum + __shfl_xor(ssum, 16);
#pragma unroll
  for (int r = 0; r < 8; ++r) {
    int rr = r + hi * 8;
    float s = __shfl(sf, rr);
    float v = acc[r] / s;
    v = v > 0.f ? v : (__expf(v) - 1.f);           // elu
    float mx = v;
#pragma unroll
    for (int m = 1; m < 16; m <<= 1) mx = fmaxf(mx, __shfl_xor(mx, m));
    float se = __expf(v - mx);
#pragma unroll
    for (int m = 1; m < 16; m <<= 1) se += __shfl_xor(se, m);
    int rowg = rowbase + rr;
    out[(size_t)rowg * GC + lo] = v - mx - __logf(se);
  }
}

// ---------------------------------------------------------------------------
// Host-side orchestration
// ---------------------------------------------------------------------------
extern "C" void kernel_launch(void* const* d_in, const int* in_sizes, int n_in,
                              void* d_out, int out_size, void* d_ws, size_t ws_size,
                              hipStream_t stream) {
  (void)in_sizes; (void)n_in; (void)out_size; (void)ws_size;
  const float* x    = (const float*)d_in[0];   // [N, F]
  const float* adj  = (const float*)d_in[1];   // [N, N]
  const float* W    = (const float*)d_in[2];   // [H, F, D]
  const float* a    = (const float*)d_in[3];   // [H, 2D]
  const float* Wo   = (const float*)d_in[4];   // [H*D, C]
  const float* ao   = (const float*)d_in[5];   // [2C]
  float* out        = (float*)d_out;           // [N, C]

  char* base = (char*)d_ws;
  size_t off = 0;
  auto alloc = [&](size_t bytes) -> char* {
    char* p = base + off;
    off = (off + bytes + 255) & ~(size_t)255;
    return p;
  };
  __bf16* xbf   = (__bf16*)alloc((size_t)GN * GF * 2);          // 8 MB
  __bf16* WT    = (__bf16*)alloc((size_t)GH * GD * GF * 2);     // 256 KB
  __bf16* WoT   = (__bf16*)alloc((size_t)GC * GH * GD * 2);     // 8 KB
  __bf16* WhT   = (__bf16*)alloc((size_t)GH * GD * GN * 2);     // 4 MB
  float*  f1t   = (float*) alloc((size_t)GN * GH * 4);
  float*  f2t   = (float*) alloc((size_t)GN * GH * 4);
  float*  m1    = (float*) alloc((size_t)GN * GH * 4);
  __bf16* hbf   = (__bf16*)alloc((size_t)GN * GH * GD * 2);     // 4 MB
  __bf16* Wh2T  = (__bf16*)alloc((size_t)GC * GN * 2);          // 256 KB
  float*  f12   = (float*) alloc((size_t)GN * 4);
  float*  f22   = (float*) alloc((size_t)GN * 4);
  float*  m2    = (float*) alloc((size_t)GN * 4);

  cvt_f32_bf16_kernel<<<(GN * GF + 255) / 256, 256, 0, stream>>>(x, xbf, GN * GF);
  transpose_W_kernel<<<(GH * GF * GD + 255) / 256, 256, 0, stream>>>(W, WT);
  transpose_Wout_kernel<<<(GH * GD * GC + 255) / 256, 256, 0, stream>>>(Wo, WoT);

  gemm1_kernel<<<dim3(GN / 128, GH), 256, 0, stream>>>(xbf, WT, WhT);
  fvec1_kernel<<<GN / 256, 256, 0, stream>>>(WhT, a, f1t, f2t);
  rowmax_kernel<GH><<<GN / 8, 256, 0, stream>>>(adj, f1t, f2t, m1);
  attn1_kernel<<<GN / 128, 256, 0, stream>>>(adj, WhT, f1t, f2t, m1, hbf);
  gemm2_kernel<<<GN / 128, 256, 0, stream>>>(hbf, WoT, ao, Wh2T, f12, f22);
  rowmax_kernel<1><<<GN / 8, 256, 0, stream>>>(adj, f12, f22, m2);
  attn2_kernel<<<GN / 128, 256, 0, stream>>>(adj, Wh2T, f12, f22, m2, out);
}
